// OddBall_AS_82506321756419
// MI455X (gfx1250) — 8-bit matrix operand order repaired and checked by reading, compiled, not run on hardware
//
#include <hip/hip_runtime.h>
#include <math.h>

#define NN 4096
#define NNv4 (NN / 4)
#define SROW (NN + 16)

typedef __attribute__((ext_vector_type(8))) int          v8i;
typedef __attribute__((ext_vector_type(4))) unsigned int v4u;
typedef __attribute__((ext_vector_type(2))) unsigned int v2u;


__global__ __launch_bounds__(256) void ob_zero_kernel(v4u* __restrict__ p, long nvec) {
  long i = (long)blockIdx.x * 256 + threadIdx.x;
  long stride = (long)gridDim.x * 256;
  v4u z = {0u, 0u, 0u, 0u};
  for (; i < nvec; i += stride) p[i] = z;
}

__global__ __launch_bounds__(256) void ob_scatter_kernel(const int* __restrict__ tri_idx,
                                                         const float* __restrict__ tri_w,
                                                         unsigned int* __restrict__ A8d,
                                                         int m) {
  int e = blockIdx.x * 256 + threadIdx.x;
  if (e >= m) return;
  int s = tri_idx[2 * e + 0];
  int d = tri_idx[2 * e + 1];
  unsigned w = (unsigned)(int)(tri_w[e] + 0.5f) & 255u;
  unsigned idx = (unsigned)s * NN + (unsigned)d;
  atomicAdd(&A8d[idx >> 2], w << ((idx & 3u) * 8u));
  if (s != d) {
    unsigned idx2 = (unsigned)d * NN + (unsigned)s;
    atomicAdd(&A8d[idx2 >> 2], w << ((idx2 & 3u) * 8u));
  }
}

__device__ __forceinline__ float ob_block_reduce(float v, float* sbuf) {
  int t = threadIdx.x;
  sbuf[t] = v;
  __syncthreads();
  for (int s = 128; s > 0; s >>= 1) {
    if (t < s) sbuf[t] += sbuf[t + s];
    __syncthreads();
  }
  float r = sbuf[0];
  __syncthreads();
  return r;
}

__global__ __launch_bounds__(256) void ob_rowsum_kernel(const unsigned int* __restrict__ A8d,
                                                        float* __restrict__ Nf) {
  __shared__ float sbuf[256];
  int row = blockIdx.x, t = threadIdx.x;
  const unsigned int* p = A8d + (long)row * NNv4;
  unsigned s = 0;
  for (int k = t; k < NNv4; k += 256) {
    unsigned v = p[k];
    s += (v & 255u) + ((v >> 8) & 255u) + ((v >> 16) & 255u) + (v >> 24);
  }
  float r = ob_block_reduce((float)s, sbuf);
  if (t == 0) Nf[row] = r;
}

__device__ __forceinline__ v8i ob_load_afrag(const unsigned char* aRow, int j) {
  v2u a0 = *(const v2u*)(aRow + j);
  v2u a1 = *(const v2u*)(aRow + j + 8);
  v2u a2 = *(const v2u*)(aRow + j + 32);
  v2u a3 = *(const v2u*)(aRow + j + 40);
  v8i af;
  af[0] = (int)a0.x; af[1] = (int)a0.y; af[2] = (int)a1.x; af[3] = (int)a1.y;
  af[4] = (int)a2.x; af[5] = (int)a2.y; af[6] = (int)a3.x; af[7] = (int)a3.y;
  return af;
}

#define OB_PACK8(dst, q0, q1)                                                   \
  dst[0] = (int)(q0).x; dst[1] = (int)(q0).y; dst[2] = (int)(q0).z;             \
  dst[3] = (int)(q0).w; dst[4] = (int)(q1).x; dst[5] = (int)(q1).y;             \
  dst[6] = (int)(q1).z; dst[7] = (int)(q1).w;

#define OB_WMMA(C, AF, BF) \
  C = __builtin_amdgcn_wmma_i32_16x16x64_iu8(false, AF, false, BF, C, false, false)

#define OB_LOADB(Q0, Q1, OFF)                                                   \
  Q0 = *(const v4u*)(bBase + (OFF));                                            \
  Q1 = *(const v4u*)(bBase + (OFF) + 32);

#define OB_PAIR(AF0, AF1, Q0, Q1, C0, C1) {                                     \
  v8i bf_; OB_PACK8(bf_, Q0, Q1);                                               \
  OB_WMMA(C0, AF0, bf_); OB_WMMA(C1, AF1, bf_); }

#define OB_JSTEP(AF0, AF1, J)                                                   \
  OB_LOADB(by0, by1, (J) + 16 * NN);                                            \
  OB_PAIR(AF0, AF1, bx0, bx1, c00, c10);                                        \
  OB_LOADB(bx0, bx1, (J) + 32 * NN);                                            \
  OB_PAIR(AF0, AF1, by0, by1, c01, c11);                                        \
  OB_LOADB(by0, by1, (J) + 48 * NN);                                            \
  OB_PAIR(AF0, AF1, bx0, bx1, c02, c12);                                        \
  OB_LOADB(bx0, bx1, (J) + 64);                                                 \
  OB_PAIR(AF0, AF1, by0, by1, c03, c13);

__global__ __launch_bounds__(256) void ob_wmma_diag_kernel(const unsigned char* __restrict__ A8,
                                                           int* __restrict__ diag3) {
  extern __shared__ unsigned char sA[];
  const int iT      = blockIdx.y;
  const int kGstart = iT >> 1;
  if ((int)blockIdx.x * 8 + 7 < kGstart) return;

  const int tid  = threadIdx.x;
  const int lane = tid & 31;
  const int wib  = tid >> 5;
  const int kG   = blockIdx.x * 8 + wib;
  const int i0   = iT * 32;
  const int k0   = kG * 64;
  const int half = lane >> 4;
  const int lm   = lane & 15;

  {
    const unsigned char* g = A8 + (long)i0 * NN;
    for (int v = tid; v < 32 * (NN / 16); v += 256) {
      int mrow = v >> 8;
      int c    = (v & 255) * 16;
      unsigned lds = (unsigned)(mrow * SROW + c);
      unsigned long long ga = (unsigned long long)(g + (long)v * 16);
      asm volatile("global_load_async_to_lds_b128 %0, %1, off"
                   :: "v"(lds), "v"(ga) : "memory");
    }
    asm volatile("s_wait_asynccnt 0" ::: "memory");
  }
  __syncthreads();

  if (kG < kGstart) return;

  const unsigned char* aRow0 = sA + lm * SROW + half * 16;
  const unsigned char* aRow1 = sA + (16 + lm) * SROW + half * 16;
  const unsigned char* bBase = A8 + (long)(k0 + lm) * NN + half * 16;

  v8i c00 = {}, c01 = {}, c02 = {}, c03 = {};
  v8i c10 = {}, c11 = {}, c12 = {}, c13 = {};

  v8i afx0 = ob_load_afrag(aRow0, 0);
  v8i afx1 = ob_load_afrag(aRow1, 0);
  v4u bx0, bx1, by0, by1;
  OB_LOADB(bx0, bx1, 0);

  for (int j = 0; j < NN; j += 128) {
    v8i afy0 = ob_load_afrag(aRow0, j + 64);
    v8i afy1 = ob_load_afrag(aRow1, j + 64);
    OB_JSTEP(afx0, afx1, j);
    afx0 = ob_load_afrag(aRow0, j + 128);
    afx1 = ob_load_afrag(aRow1, j + 128);
    OB_JSTEP(afy0, afy1, j + 64);
  }

  int col00 = 0, col01 = 0, col02 = 0, col03 = 0;
  int col10 = 0, col11 = 0, col12 = 0, col13 = 0;
  const int ia0 = i0, ia1 = i0 + 16;

#pragma unroll
  for (int r = 0; r < 8; ++r) {
    int m2 = r + 8 * half;
    {
      const unsigned char* pr = sA + m2 * SROW + k0 + lm;
      int p0 = (int)pr[0]  * c00[r];
      int p1 = (int)pr[16] * c01[r];
      int p2 = (int)pr[32] * c02[r];
      int p3 = (int)pr[48] * c03[r];
      col00 += p0; col01 += p1; col02 += p2; col03 += p3;
      int acc = (k0      >= ia0 ? p0 : 0) + (k0 + 16 >= ia0 ? p1 : 0)
              + (k0 + 32 >= ia0 ? p2 : 0) + (k0 + 48 >= ia0 ? p3 : 0);
      acc += __shfl_xor(acc, 1); acc += __shfl_xor(acc, 2);
      acc += __shfl_xor(acc, 4); acc += __shfl_xor(acc, 8);
      if (lm == 0) atomicAdd(&diag3[i0 + m2], acc);
    }
    {
      const unsigned char* pr = sA + (16 + m2) * SROW + k0 + lm;
      int p0 = (int)pr[0]  * c10[r];
      int p1 = (int)pr[16] * c11[r];
      int p2 = (int)pr[32] * c12[r];
      int p3 = (int)pr[48] * c13[r];
      col10 += p0; col11 += p1; col12 += p2; col13 += p3;
      int acc = (k0      >= ia1 ? p0 : 0) + (k0 + 16 >= ia1 ? p1 : 0)
              + (k0 + 32 >= ia1 ? p2 : 0) + (k0 + 48 >= ia1 ? p3 : 0);
      acc += __shfl_xor(acc, 1); acc += __shfl_xor(acc, 2);
      acc += __shfl_xor(acc, 4); acc += __shfl_xor(acc, 8);
      if (lm == 0) atomicAdd(&diag3[i0 + 16 + m2], acc);
    }
  }

  col00 += __shfl_xor(col00, 16); col01 += __shfl_xor(col01, 16);
  col02 += __shfl_xor(col02, 16); col03 += __shfl_xor(col03, 16);
  col10 += __shfl_xor(col10, 16); col11 += __shfl_xor(col11, 16);
  col12 += __shfl_xor(col12, 16); col13 += __shfl_xor(col13, 16);
  if (half == 0) {
    if (k0      > ia0) atomicAdd(&diag3[k0      + lm], col00);
    if (k0 + 16 > ia0) atomicAdd(&diag3[k0 + 16 + lm], col01);
    if (k0 + 32 > ia0) atomicAdd(&diag3[k0 + 32 + lm], col02);
    if (k0 + 48 > ia0) atomicAdd(&diag3[k0 + 48 + lm], col03);
    if (k0      > ia1) atomicAdd(&diag3[k0      + lm], col10);
    if (k0 + 16 > ia1) atomicAdd(&diag3[k0 + 16 + lm], col11);
    if (k0 + 32 > ia1) atomicAdd(&diag3[k0 + 32 + lm], col12);
    if (k0 + 48 > ia1) atomicAdd(&diag3[k0 + 48 + lm], col13);
  }
}

__global__ __launch_bounds__(256) void ob_finalize_kernel(const float* __restrict__ Nf,
                                                          const int* __restrict__ diag3,
                                                          float* __restrict__ Ef,
                                                          const int* __restrict__ lst,
                                                          int nT,
                                                          float* __restrict__ out) {
  __shared__ float sbuf[256];
  __shared__ float sh_b, sh_w;
  int t = threadIdx.x;
  float s1 = 0.f, s2 = 0.f, sy = 0.f, sxy = 0.f;
  for (int i = t; i < NN; i += 256) {
    float Nv = Nf[i];
    float Ev = Nv + 0.5f * (float)diag3[i];
    Ef[i] = Ev;
    float x = logf(Nv + 1e-20f);
    float y = logf(Ev + 1e-20f);
    s1 += x; s2 += x * x; sy += y; sxy += x * y;
  }
  float S1  = ob_block_reduce(s1, sbuf);
  float S2  = ob_block_reduce(s2, sbuf);
  float SY  = ob_block_reduce(sy, sbuf);
  float SXY = ob_block_reduce(sxy, sbuf);
  if (t == 0) {
    float n = (float)NN;
    float det = n * S2 - S1 * S1;
    float ww = (n * SXY - S1 * SY) / det;
    float bb = (SY - ww * S1) / n;
    sh_w = ww; sh_b = bb;
  }
  __syncthreads();
  float eb = expf(sh_b), ww = sh_w;
  float acc = 0.f;
  for (int k = t; k < nT; k += 256) {
    int l = lst[k];
    float r = eb * powf(Nf[l], ww) - Ef[l];
    acc += r * r;
  }
  float tot = ob_block_reduce(acc, sbuf);
  if (t == 0) out[0] = tot;
}


extern "C" void kernel_launch(void* const* d_in, const int* in_sizes, int n_in,
                              void* d_out, int out_size, void* d_ws, size_t ws_size,
                              hipStream_t stream) {
  const int*   tri_idx = (const int*)d_in[0];
  const float* tri_w   = (const float*)d_in[1];
  const int*   lst     = (const int*)d_in[2];
  const int m  = in_sizes[1];
  const int nT = in_sizes[2];

  unsigned char* A8    = (unsigned char*)d_ws;
  int*           diag3 = (int*)(A8 + (long)NN * NN);
  float*         Nf    = (float*)(diag3 + NN);
  float*         Ef    = Nf + NN;

  long nvec = ((long)NN * NN + (long)NN * sizeof(int)) / 16;
  ob_zero_kernel<<<2048, 256, 0, stream>>>((v4u*)d_ws, nvec);

  ob_scatter_kernel<<<(m + 255) / 256, 256, 0, stream>>>(tri_idx, tri_w,
                                                         (unsigned int*)A8, m);

  ob_rowsum_kernel<<<NN, 256, 0, stream>>>((const unsigned int*)A8, Nf);

  ob_wmma_diag_kernel<<<dim3(8, 128), 256, 32 * SROW + 64, stream>>>(A8, diag3);

  ob_finalize_kernel<<<1, 256, 0, stream>>>(Nf, diag3, Ef, lst, nT, (float*)d_out);
}
